// UResNetEncoder_41566693491195
// MI455X (gfx1250) — matrix operand order repaired and checked by reading, not compiled, not run on hardware
//
#include <hip/hip_runtime.h>
#include <cstdint>
#include <cstddef>

// ---------------------------------------------------------------------------
// CDNA5 (gfx1250) UResNet encoder: implicit-GEMM conv via V_WMMA_F32_16X16X32_BF16
// wave32, 16x16 C/D tiles, f32 accumulate, fused BN-stats reduction.
// All conv shapes (Cin/Cout/k/stride/pad AND spatial dims) are compile-time:
// every (NT,KSZ) combo maps to exactly one pyramid level, so DOUT = 128>>(NT-1)
// is a power of two -> voxel decomposition is shifts/masks, zero runtime divs.
// ---------------------------------------------------------------------------

typedef __attribute__((ext_vector_type(16))) __bf16 v16bf;
typedef __attribute__((ext_vector_type(2)))  __bf16 v2bf;
typedef __attribute__((ext_vector_type(8)))  float  v8f;

__device__ __forceinline__ unsigned int pk_bf16(float a, float b) {
  // native fptrunc f32->bf16 (RNE); backend selects hardware cvt on gfx1250
  union { v2bf h; unsigned int u; } r;
  r.h[0] = (__bf16)a;
  r.h[1] = (__bf16)b;
  return r.u;
}

// ---------------------------------------------------------------------------
// Weight packer: fp32 [k^3][Cin][Cout] (DHWIO flat = K-major, K = tap*Cin+ci)
//   -> chunk-major dword-packed bf16 pairs:
//      WQ[(chunk32 * Cout + n) * 16 + k2rel],  K = chunk32*32 + k2rel*2
// so each lane's 8 B-fragment dwords are contiguous (2 x b128 loads).
// ---------------------------------------------------------------------------
__global__ void pack_weights_kernel(const float* __restrict__ W,
                                    unsigned int* __restrict__ WQ,
                                    int Ktot, int Cout, int total) {
  int tid = blockIdx.x * blockDim.x + threadIdx.x;
  if (tid >= total) return;
  int k2rel = tid & 15;
  int t2    = tid >> 4;
  int n     = t2 % Cout;
  int ch    = t2 / Cout;
  int k0    = ch * 32 + k2rel * 2;
  float f0 = (k0     < Ktot) ? W[(size_t)k0 * Cout + n]       : 0.f;
  float f1 = (k0 + 1 < Ktot) ? W[(size_t)(k0 + 1) * Cout + n] : 0.f;
  WQ[tid] = pk_bf16(f0, f1);
}

// ---------------------------------------------------------------------------
// Implicit-GEMM conv:  Y[m, n] = sum_K A[m, K] * Wq[K, n]
//   m = output voxel (DOUT^3, row-major d,h,w),  n = Cout,  K = tap*Cin + ci
// One workgroup = 128 threads = 4 waves; each wave owns a 16-row M tile and
// all NT (= Cout/16) n-tiles.  bf16 WMMA 16x16x32, f32 accumulators.
// Per-channel sum / sum-of-squares accumulated for the following BatchNorm
// (C/D layout: N depends only on lane, so each lane reduces one channel).
// ---------------------------------------------------------------------------
template <int NT, int CINB, int KSZ>
__global__ __launch_bounds__(128)
void conv_wmma_kernel(const float* __restrict__ X,
                      const unsigned int* __restrict__ WQ,
                      float* __restrict__ Y,
                      float* __restrict__ st) {  // [2*Cout] sum,sumsq (or null)
  constexpr int COUT   = NT * 16;
  constexpr int CIN    = CINB * 16;
  constexpr int KTOT   = KSZ * KSZ * KSZ * CIN;
  constexpr int KPAD   = (KTOT + 31) & ~31;
  constexpr int STRIDE = (KSZ == 3) ? 1 : 2;
  constexpr int PAD    = (KSZ == 3) ? 1 : 0;
  constexpr int DOUT   = 128 >> (NT - 1);             // level = NT-1
  constexpr int DIN    = (KSZ == 3) ? DOUT : 2 * DOUT;
  constexpr int LB     = (NT == 1) ? 7 : (NT == 2) ? 6 : (NT == 3) ? 5
                       : (NT == 4) ? 4 : 3;           // log2(DOUT)
  constexpr int DMASK  = DOUT - 1;

  const int lane  = threadIdx.x & 31;
  const int wv    = threadIdx.x >> 5;
  const int mbase = (blockIdx.x * 4 + wv) * 16;
  const int row   = lane & 15;
  const int m     = mbase + row;

  // voxel coords of this lane's A row (pure shifts/masks)
  const int wo = m & DMASK;
  const int ho = (m >> LB) & DMASK;
  const int dn = m >> (2 * LB);
  const int bz = dn * STRIDE - PAD;
  const int by = ho * STRIDE - PAD;
  const int bx = wo * STRIDE - PAD;
  const int hi = (lane & 16) ? 8 : 0;   // K-half selector for A and B layouts

  v8f acc[NT];
#pragma unroll
  for (int t = 0; t < NT; ++t)
#pragma unroll
    for (int e = 0; e < 8; ++e) acc[t][e] = 0.f;

  for (int kc = 0; kc < KPAD; kc += 32) {
    // ---- build A fragment (16x32 bf16): lane holds row M=lane&15,
    //      VGPR j holds K pair base = 2*hi + j*2
    union { unsigned int u[8]; v16bf v; } Af;
#pragma unroll
    for (int j = 0; j < 8; ++j) {
      int K = kc + (hi << 1) + (j << 1);
      float f0 = 0.f, f1 = 0.f;
      if (K < KTOT) {
        int tap = (unsigned)K / (unsigned)CIN;       // constant div -> mulhi
        int ci  = K - tap * CIN;                     // even (CIN even)
        int kd  = tap / (KSZ * KSZ);
        int rr  = tap - kd * (KSZ * KSZ);
        int kh  = rr / KSZ;
        int kw  = rr - kh * KSZ;
        int z = bz + kd, y = by + kh, x = bx + kw;
        if ((unsigned)z < (unsigned)DIN && (unsigned)y < (unsigned)DIN &&
            (unsigned)x < (unsigned)DIN) {
          const float2 f2 = *(const float2*)(
              X + (((size_t)z * DIN + y) * (size_t)DIN + x) * (size_t)CIN + ci);
          f0 = f2.x;
          f1 = f2.y;
        }
      }
      Af.u[j] = pk_bf16(f0, f1);
    }

    // ---- B fragments: contiguous chunk-major packed weights, 2 x b128 each
    const size_t chunkBase = (size_t)(kc >> 5) * COUT * 16;
    if (kc + 32 < KPAD)  // prefetch next chunk's B block (global_prefetch_b8)
      __builtin_prefetch(WQ + chunkBase + (size_t)COUT * 16 + (row << 4), 0, 0);
#pragma unroll
    for (int t = 0; t < NT; ++t) {
      union { uint4 q[2]; v16bf v; } Bf;
      const uint4* bq = (const uint4*)(WQ + chunkBase +
                                       (size_t)(t * 16 + row) * 16 + hi);
      Bf.q[0] = bq[0];
      Bf.q[1] = bq[1];
      acc[t] = __builtin_amdgcn_wmma_f32_16x16x32_bf16(
          false, Af.v, false, Bf.v, (short)0, acc[t], false, false);
    }
  }

  // ---- epilogue: C layout (lane&15 = N, VGPR r = row r + hi) + BN stats
  const int mrow0 = mbase + hi;
#pragma unroll
  for (int t = 0; t < NT; ++t) {
    const int n = t * 16 + row;
    float s = 0.f, q = 0.f;
#pragma unroll
    for (int r = 0; r < 8; ++r) {
      float v = acc[t][r];
      Y[(size_t)(mrow0 + r) * COUT + n] = v;
      s += v;
      q += v * v;
    }
    if (st) {
      atomicAdd(&st[n], s);
      atomicAdd(&st[COUT + n], q);
    }
  }
}

// ---------------------------------------------------------------------------
// BN finalize: per-channel effective scale/shift from sum/sumsq.
// ---------------------------------------------------------------------------
__global__ void bn_finalize_kernel(const float* __restrict__ st,
                                   const float* __restrict__ gamma,
                                   const float* __restrict__ beta,
                                   float* __restrict__ ss, int C, float invN) {
  int c = threadIdx.x;
  if (c >= C) return;
  float mean = st[c] * invN;
  float var  = st[C + c] * invN - mean * mean;
  float sc   = gamma[c] * rsqrtf(var + 1e-5f);
  ss[c]      = sc;
  ss[C + c]  = beta[c] - mean * sc;
}

// ---------------------------------------------------------------------------
// Fused BN * LeakyReLU (+ optional residual, optional output-stats for the
// down-path BN).  One thread = one channel x 16 voxels -> 2 atomics max.
// C is compile-time -> constant-division indexing.
// ---------------------------------------------------------------------------
template <int C>
__global__ void bn_act_kernel(const float* __restrict__ X,
                              const float* __restrict__ ss,
                              const float* __restrict__ res,
                              float* __restrict__ Y,
                              float* __restrict__ ost, int Mt) {
  int tid = blockIdx.x * blockDim.x + threadIdx.x;
  int total = (Mt >> 4) * C;
  if (tid >= total) return;
  int c  = tid % C;
  int mb = tid / C;
  float sc = ss[c], sh = ss[C + c];
  size_t base = (size_t)mb * 16 * C + c;
  float s = 0.f, q = 0.f;
#pragma unroll 4
  for (int r = 0; r < 16; ++r) {
    size_t i = base + (size_t)r * C;
    float v = X[i] * sc + sh;
    if (res) v += res[i];
    v = (v >= 0.f) ? v : 0.01f * v;
    Y[i] = v;
    s += v;
    q += v * v;
  }
  if (ost) {
    atomicAdd(&ost[c], s);
    atomicAdd(&ost[C + c], q);
  }
}

// ---------------------------------------------------------------------------
// Stem conv: Cin = 1, 3x3x3 SAME, Cout = 16 (tiny: ~1.8 GFLOP, VALU is fine).
// ---------------------------------------------------------------------------
__global__ void input_conv_kernel(const float* __restrict__ X,
                                  const float* __restrict__ W,
                                  float* __restrict__ Y) {
  constexpr int Dn = 128;
  int m = blockIdx.x * blockDim.x + threadIdx.x;
  int x = m & 127;
  int y = (m >> 7) & 127;
  int z = m >> 14;
  float xv[27];
#pragma unroll
  for (int kd = 0; kd < 3; ++kd)
#pragma unroll
    for (int kh = 0; kh < 3; ++kh)
#pragma unroll
      for (int kw = 0; kw < 3; ++kw) {
        int zz = z + kd - 1, yy = y + kh - 1, xx = x + kw - 1;
        bool in = (unsigned)zz < (unsigned)Dn && (unsigned)yy < (unsigned)Dn &&
                  (unsigned)xx < (unsigned)Dn;
        xv[kd * 9 + kh * 3 + kw] =
            in ? X[((size_t)zz * Dn + yy) * Dn + xx] : 0.f;
      }
  for (int co = 0; co < 16; ++co) {
    float a = 0.f;
#pragma unroll
    for (int tt = 0; tt < 27; ++tt) a += xv[tt] * W[tt * 16 + co];
    Y[(size_t)m * 16 + co] = a;
  }
}

// ---------------------------------------------------------------------------
// Host orchestration
// ---------------------------------------------------------------------------
extern "C" void kernel_launch(void* const* d_in, const int* in_sizes, int n_in,
                              void* d_out, int out_size, void* d_ws,
                              size_t ws_size, hipStream_t stream) {
  (void)out_size; (void)ws_size;
  static const int DL[5] = {128, 64, 32, 16, 8};
  static const int FL[5] = {16, 32, 48, 64, 80};

  auto F32 = [&](int i) { return (const float*)d_in[i]; };

  // Input ordering: dict-insertion (x first) vs jax-pytree alphabetical (x last).
  bool alpha = !(n_in > 0 && in_sizes[0] == 128 * 128 * 128);
  const float* xin = (const float*)d_in[alpha ? (n_in - 1) : 0];
  int p = alpha ? 0 : 1;
  const float* input_w = F32(p++);
  const float *W1[5][2], *G1[5][2], *B1[5][2], *W2[5][2], *G2[5][2], *B2[5][2];
  const float *DG[4], *DB[4], *DW[4];
  for (int i = 0; i < 5; ++i) {
    for (int b = 0; b < 2; ++b) {
      if (!alpha) {
        W1[i][b]=F32(p++); G1[i][b]=F32(p++); B1[i][b]=F32(p++);
        W2[i][b]=F32(p++); G2[i][b]=F32(p++); B2[i][b]=F32(p++);
      } else { // alphabetical within block dict: b1,b2,g1,g2,w1,w2
        B1[i][b]=F32(p++); B2[i][b]=F32(p++); G1[i][b]=F32(p++);
        G2[i][b]=F32(p++); W1[i][b]=F32(p++); W2[i][b]=F32(p++);
      }
    }
    if (i < 4) {
      if (!alpha) { DG[i]=F32(p++); DB[i]=F32(p++); DW[i]=F32(p++); }
      else        { DB[i]=F32(p++); DG[i]=F32(p++); DW[i]=F32(p++); }
    }
  }

  // Workspace layout: three 134MB ping-pong buffers + packed weights + stats.
  char* ws = (char*)d_ws;
  float* A    = (float*)(ws);
  float* Bb   = (float*)(ws + 134217728);
  float* Cb   = (float*)(ws + 268435456);
  char*  Wreg = ws + 402653184;
  float* stp  = (float*)(ws + 402653184 + (4u << 20));
  float* ssp  = stp + 1024;
  float* bstp = stp + 2048;

  // Output regions (concatenated encoder tensors + final).
  float* OUT = (float*)d_out;
  float* T[7] = {OUT, OUT + 33554432, OUT + 67108864, OUT + 75497472,
                 OUT + 77070336, OUT + 77332480, OUT + 77373440};
  float* encOut[5] = {T[1], T[2], T[3], T[4], T[5]};

  // ---- pack all conv weights to chunk-major bf16 WMMA-B layout
  size_t woff = 0;
  auto packW = [&](const float* w, int Ktot, int Cout) -> const unsigned int* {
    int Kpad = (Ktot + 31) & ~31;
    unsigned int* dst = (unsigned int*)(Wreg + woff);
    int total = (Kpad / 32) * Cout * 16;
    woff += (size_t)total * 4;
    pack_weights_kernel<<<(total + 255) / 256, 256, 0, stream>>>(w, dst, Ktot,
                                                                 Cout, total);
    return dst;
  };
  const unsigned int *Q1[5][2], *Q2[5][2], *QD[4];
  for (int i = 0; i < 5; ++i)
    for (int b = 0; b < 2; ++b) {
      Q1[i][b] = packW(W1[i][b], 27 * FL[i], FL[i]);
      Q2[i][b] = packW(W2[i][b], 27 * FL[i], FL[i]);
    }
  for (int i = 0; i < 4; ++i) QD[i] = packW(DW[i], 8 * FL[i], FL[i + 1]);

  // ---- conv dispatch: every shape fully compile-time.
  //      k=3 (s=1,p=1): Cin == Cout == nt*16, DOUT = 128>>(nt-1)
  //      k=2 (s=2,p=0): Cin == (nt-1)*16,     DOUT = 128>>(nt-1)
  auto conv = [&](const float* X, const unsigned int* WQ, float* Y, float* st,
                  int Dout, int Cout, int ksz) {
    int Mt = Dout * Dout * Dout;
    dim3 grid(Mt / 64);
    int nt = Cout >> 4;
    if (ksz == 3) {
      switch (nt) {
        case 1: conv_wmma_kernel<1,1,3><<<grid,128,0,stream>>>(X,WQ,Y,st); break;
        case 2: conv_wmma_kernel<2,2,3><<<grid,128,0,stream>>>(X,WQ,Y,st); break;
        case 3: conv_wmma_kernel<3,3,3><<<grid,128,0,stream>>>(X,WQ,Y,st); break;
        case 4: conv_wmma_kernel<4,4,3><<<grid,128,0,stream>>>(X,WQ,Y,st); break;
        default: conv_wmma_kernel<5,5,3><<<grid,128,0,stream>>>(X,WQ,Y,st); break;
      }
    } else {
      switch (nt) {
        case 2: conv_wmma_kernel<2,1,2><<<grid,128,0,stream>>>(X,WQ,Y,st); break;
        case 3: conv_wmma_kernel<3,2,2><<<grid,128,0,stream>>>(X,WQ,Y,st); break;
        case 4: conv_wmma_kernel<4,3,2><<<grid,128,0,stream>>>(X,WQ,Y,st); break;
        default: conv_wmma_kernel<5,4,2><<<grid,128,0,stream>>>(X,WQ,Y,st); break;
      }
    }
  };
  auto zero_st = [&](float* s) {
    hipMemsetAsync(s, 0, 2 * 128 * sizeof(float), stream);
  };
  auto act = [&](const float* X, const float* res, float* Y, float* ost,
                 int Mt, int C) {
    int total = (Mt >> 4) * C;
    dim3 grid((total + 255) / 256);
    switch (C >> 4) {
      case 1: bn_act_kernel<16><<<grid,256,0,stream>>>(X, ssp, res, Y, ost, Mt); break;
      case 2: bn_act_kernel<32><<<grid,256,0,stream>>>(X, ssp, res, Y, ost, Mt); break;
      case 3: bn_act_kernel<48><<<grid,256,0,stream>>>(X, ssp, res, Y, ost, Mt); break;
      case 4: bn_act_kernel<64><<<grid,256,0,stream>>>(X, ssp, res, Y, ost, Mt); break;
      default: bn_act_kernel<80><<<grid,256,0,stream>>>(X, ssp, res, Y, ost, Mt); break;
    }
  };

  // ---- stem conv -> encoder tensor 0
  input_conv_kernel<<<(128 * 128 * 128) / 256, 256, 0, stream>>>(xin, input_w,
                                                                 T[0]);

  float* xcur = T[0];
  for (int i = 0; i < 5; ++i) {
    int D = DL[i], Fc = FL[i];
    int Mt = D * D * D;
    float invN = 1.f / (float)Mt;
    for (int b = 0; b < 2; ++b) {
      bool last = (b == 1);
      float* dest = last ? encOut[i] : Cb;  // (level>0: in-place over Cb, safe)
      // conv1 -> BN -> lrelu
      zero_st(stp);
      conv(xcur, Q1[i][b], A, stp, D, Fc, 3);
      bn_finalize_kernel<<<1, 128, 0, stream>>>(stp, G1[i][b], B1[i][b], ssp,
                                                Fc, invN);
      act(A, nullptr, Bb, nullptr, Mt, Fc);
      // conv2 -> BN -> +residual -> lrelu
      zero_st(stp);
      conv(Bb, Q2[i][b], A, stp, D, Fc, 3);
      bn_finalize_kernel<<<1, 128, 0, stream>>>(stp, G2[i][b], B2[i][b], ssp,
                                                Fc, invN);
      bool needStats = last && (i < 4);
      if (needStats) zero_st(bstp);
      act(A, xcur, dest, needStats ? bstp : nullptr, Mt, Fc);
      xcur = dest;
    }
    if (i < 4) {  // down: BN -> lrelu -> conv(k=2, s=2)
      bn_finalize_kernel<<<1, 128, 0, stream>>>(bstp, DG[i], DB[i], ssp, Fc,
                                                invN);
      act(xcur, nullptr, Bb, nullptr, Mt, Fc);
      conv(Bb, QD[i], Cb, nullptr, D / 2, FL[i + 1], 2);
      xcur = Cb;
    }
  }
  // final == last encoder tensor
  hipMemcpyAsync(T[6], T[5], (size_t)8 * 8 * 8 * 80 * sizeof(float),
                 hipMemcpyDeviceToDevice, stream);
}
